// TransLayer_80719615361794
// MI455X (gfx1250) — matrix operand order repaired and checked by reading, not compiled, not run on hardware
//
#include <hip/hip_runtime.h>
#include <cstdint>

typedef uint16_t u16;
typedef __attribute__((ext_vector_type(16))) __bf16 v16bf;
typedef __attribute__((ext_vector_type(8)))  float  v8f;
typedef __attribute__((ext_vector_type(2)))  float  v2f;

// ---------------- bf16 <-> f32 helpers (native converts) ----------------------
__device__ __forceinline__ u16 f2bf(float f) {
  return __builtin_bit_cast(u16, (__bf16)f);
}
__device__ __forceinline__ float bf2f(u16 h) {
  return (float)__builtin_bit_cast(__bf16, h);
}
__device__ __forceinline__ uint32_t pack_pair(uint32_t a, uint32_t b, int hi) {
  return hi ? ((a >> 16) | (b & 0xffff0000u)) : ((a & 0xffffu) | (b << 16));
}
__device__ __forceinline__ float sum8bf(uint4 u) {
  float s = 0.f;
  s += bf2f((u16)(u.x)) + bf2f((u16)(u.x >> 16));
  s += bf2f((u16)(u.y)) + bf2f((u16)(u.y >> 16));
  s += bf2f((u16)(u.z)) + bf2f((u16)(u.z >> 16));
  s += bf2f((u16)(u.w)) + bf2f((u16)(u.w >> 16));
  return s;
}
__device__ __forceinline__ v16bf frag16(uint4 a, uint4 b) {
  union { uint4 u[2]; v16bf v; } t; t.u[0] = a; t.u[1] = b; return t.v;
}

// ---------------- WMMA fragment loaders (ISA 7.12.2, wave32) ------------------
__device__ __forceinline__ v16bf load_a_bf16(const u16* a, int lda, int lane) {
  int r = lane & 15, h = (lane >> 4) & 1;
  const u16* row = a + r * lda + 8 * h;
  return frag16(*(const uint4*)row, *(const uint4*)(row + 16));
}
__device__ __forceinline__ v16bf load_bt_bf16(const u16* m, int ldm, int lane) {
  int n = lane & 15, h = (lane >> 4) & 1;
  const u16* row = m + n * ldm + 8 * h;
  return frag16(*(const uint4*)row, *(const uint4*)(row + 16));
}
__device__ __forceinline__ v2f load_a_f32(const float* a, int lda, int lane) {
  int r = lane & 15, h = (lane >> 4) & 1;
  const float* p = a + r * lda + 2 * h;
  v2f o; o[0] = p[0]; o[1] = p[1]; return o;
}
__device__ __forceinline__ v2f load_bt_f32(const float* m, int ldm, int lane) {
  int n = lane & 15, h = (lane >> 4) & 1;
  const float* p = m + n * ldm + 2 * h;
  v2f o; o[0] = p[0]; o[1] = p[1]; return o;
}

__device__ __forceinline__ v8f wmma_bf16(v16bf a, v16bf b, v8f c) {
  return __builtin_amdgcn_wmma_f32_16x16x32_bf16(false, a, false, b, (short)0, c, false, false);
}
__device__ __forceinline__ v8f wmma_f32(v2f a, v2f b, v8f c) {
  return __builtin_amdgcn_wmma_f32_16x16x4_f32(false, a, false, b, (short)0, c, false, false);
}

// ---------------- CDNA5 async global->LDS DMA (ASYNCcnt-tracked) --------------
__device__ __forceinline__ void async_ld_b128(uint32_t lds_off, const void* gptr) {
  asm volatile("global_load_async_to_lds_b128 %0, %1, off"
               :: "v"(lds_off), "v"((uint64_t)(uintptr_t)gptr) : "memory");
}
__device__ __forceinline__ void wait_async0() {
  asm volatile("s_wait_asynccnt 0" ::: "memory");
}
// Contiguous tile copy: TOT 16B chunks (multiple of 256), fully unrolled.
template <int TOT>
__device__ __forceinline__ void async_copy_contig(void* lds, const void* g, int tid) {
  uint32_t base = (uint32_t)(uintptr_t)lds;
#pragma unroll
  for (int it = 0; it < TOT / 256; ++it) {
    int idx = tid + it * 256;
    async_ld_b128(base + idx * 16, (const char*)g + (size_t)idx * 16);
  }
}
// Row-strided tile copy: ROWS rows of CPR 16B chunks, packed row-major in LDS.
template <int ROWS, int CPR>
__device__ __forceinline__ void async_copy_rows(void* lds, const char* g, size_t gstride, int tid) {
  uint32_t base = (uint32_t)(uintptr_t)lds;
#pragma unroll
  for (int it = 0; it < (ROWS * CPR) / 256; ++it) {
    int idx = tid + it * 256;
    int r = idx / CPR, c = idx % CPR;
    async_ld_b128(base + idx * 16, g + (size_t)r * gstride + (size_t)c * 16);
  }
}

// =============================================================================
// K1: weights -> bf16 (packed b32 stores)
__global__ void k_cvt_weights(const float* wqkv, const float* wout, u16* wqkvb, u16* woutb) {
  int i = blockIdx.x * 256 + threadIdx.x;
  int i2 = i * 2;
  if (i2 < 512 * 1536) {
    uint32_t pk = (uint32_t)f2bf(wqkv[i2]) | ((uint32_t)f2bf(wqkv[i2 + 1]) << 16);
    *(uint32_t*)(wqkvb + i2) = pk;
  }
  if (i2 < 512 * 512) {
    uint32_t pk = (uint32_t)f2bf(wout[i2]) | ((uint32_t)f2bf(wout[i2 + 1]) << 16);
    *(uint32_t*)(woutb + i2) = pk;
  }
}

// K2: LayerNorm, one block per row of 512; packed bf16 stores.
__global__ void k_layernorm(const float* x, const float* g, const float* b, u16* xn) {
  int row = blockIdx.x;
  const float* xr = x + (size_t)row * 512;
  int t = threadIdx.x;
  float2 xv = *(const float2*)(xr + 2 * t);
  float a0 = xv.x, a1 = xv.y;
  __shared__ float red[256];
  red[t] = a0 + a1; __syncthreads();
  for (int o = 128; o > 0; o >>= 1) { if (t < o) red[t] += red[t + o]; __syncthreads(); }
  float mean = red[0] * (1.f / 512.f); __syncthreads();
  red[t] = a0 * a0 + a1 * a1; __syncthreads();
  for (int o = 128; o > 0; o >>= 1) { if (t < o) red[t] += red[t + o]; __syncthreads(); }
  float var = red[0] * (1.f / 512.f) - mean * mean;
  float rs = rsqrtf(var + 1e-5f);
  float y0 = (a0 - mean) * rs * g[2 * t] + b[2 * t];
  float y1 = (a1 - mean) * rs * g[2 * t + 1] + b[2 * t + 1];
  uint32_t pk = (uint32_t)f2bf(y0) | ((uint32_t)f2bf(y1) << 16);
  *(uint32_t*)(xn + (size_t)row * 512 + 2 * t) = pk;
}

// K3: QKV GEMM, bf16 WMMA. A tile async-DMA'd; B transpose-staged (stride 40).
__global__ void k_gemm_qkv(const u16* XN, const u16* Wb, u16* Qb, u16* Kb, u16* Vb) {
  __shared__ __align__(16) u16 As[128 * 32];
  __shared__ __align__(16) u16 Bst[128 * 40];
  int gm0 = blockIdx.y * 128, gn0 = blockIdx.x * 128;
  int tid = threadIdx.x, lane = tid & 31, w = tid >> 5;
  int wm = (w & 3) * 32, wn = (w >> 2) * 64;
  v8f acc[2][4] = {};
  for (int kk = 0; kk < 512; kk += 32) {
    async_copy_rows<128, 4>(As, (const char*)(XN + (size_t)gm0 * 512 + kk), 1024, tid);
    {  // B: 32x128, two k-rows packed per b32, transposed store
      int kp = tid >> 4, ng = tid & 15;
      const u16* g0 = Wb + (size_t)(kk + 2 * kp) * 1536 + gn0 + ng * 8;
      uint4 ra = *(const uint4*)g0, rb = *(const uint4*)(g0 + 1536);
      uint32_t av[4] = {ra.x, ra.y, ra.z, ra.w}, bv[4] = {rb.x, rb.y, rb.z, rb.w};
#pragma unroll
      for (int nn = 0; nn < 8; ++nn)
        *(uint32_t*)(Bst + (ng * 8 + nn) * 40 + 2 * kp) = pack_pair(av[nn >> 1], bv[nn >> 1], nn & 1);
    }
    wait_async0();
    __syncthreads();
    v16bf a0 = load_a_bf16(As + (wm + 0) * 32, 32, lane);
    v16bf a1 = load_a_bf16(As + (wm + 16) * 32, 32, lane);
#pragma unroll
    for (int j = 0; j < 4; ++j) {
      v16bf b = load_bt_bf16(Bst + (wn + j * 16) * 40, 40, lane);
      acc[0][j] = wmma_bf16(a0, b, acc[0][j]);
      acc[1][j] = wmma_bf16(a1, b, acc[1][j]);
    }
    __syncthreads();
  }
  int col = lane & 15, h = (lane >> 4) & 1;
#pragma unroll
  for (int i = 0; i < 2; ++i)
#pragma unroll
    for (int j = 0; j < 4; ++j) {
      int gc = gn0 + wn + j * 16 + col;
      int which = gc >> 9, hd = gc & 511, hh = hd >> 6, d = hd & 63;
      u16* dst = which == 0 ? Qb : (which == 1 ? Kb : Vb);
      float sc = which == 0 ? 0.125f : 1.0f;  // SCALE = 64^-0.5
#pragma unroll
      for (int e = 0; e < 8; ++e) {
        int gr = gm0 + wm + i * 16 + 8 * h + e;
        int bb = gr >> 13, n = gr & 8191;
        dst[(((size_t)bb * 8 + hh) * 8192 + n) * 64 + d] = f2bf(sc * acc[i][j][e]);
      }
    }
}

// K4: landmark means over l=32 chunks (b128 loads + LDS reduce).
__global__ void k_landmarks(const u16* Qb, const u16* Kb, float* QLf, float* KLf, u16* QLb, u16* KLb) {
  int bh = blockIdx.x >> 8, j = blockIdx.x & 255;
  int t = threadIdx.x;          // 64 threads
  int dg = t & 7, rg = t >> 3;  // 8 dim-groups x 8 row-groups (4 rows each)
  const u16* qp = Qb + ((size_t)bh * 8192 + j * 32 + rg * 4) * 64 + dg * 8;
  const u16* kp = Kb + ((size_t)bh * 8192 + j * 32 + rg * 4) * 64 + dg * 8;
  float sq[8] = {}, sk[8] = {};
#pragma unroll
  for (int r = 0; r < 4; ++r) {
    uint4 qv = *(const uint4*)(qp + r * 64);
    uint4 kv = *(const uint4*)(kp + r * 64);
    uint32_t qa[4] = {qv.x, qv.y, qv.z, qv.w}, ka[4] = {kv.x, kv.y, kv.z, kv.w};
#pragma unroll
    for (int q = 0; q < 4; ++q) {
      sq[2 * q] += bf2f((u16)(qa[q] & 0xffff)); sq[2 * q + 1] += bf2f((u16)(qa[q] >> 16));
      sk[2 * q] += bf2f((u16)(ka[q] & 0xffff)); sk[2 * q + 1] += bf2f((u16)(ka[q] >> 16));
    }
  }
  __shared__ float rq[8][64], rk[8][64];
#pragma unroll
  for (int q = 0; q < 8; ++q) { rq[rg][dg * 8 + q] = sq[q]; rk[rg][dg * 8 + q] = sk[q]; }
  __syncthreads();
  float fq = 0.f, fk = 0.f;
#pragma unroll
  for (int g = 0; g < 8; ++g) { fq += rq[g][t]; fk += rk[g][t]; }
  fq *= (1.f / 32.f); fk *= (1.f / 32.f);
  size_t o = ((size_t)bh * 256 + j) * 64 + t;
  QLf[o] = fq; KLf[o] = fk; QLb[o] = f2bf(fq); KLb[o] = f2bf(fk);
}

// K5a: E = exp(QL @ KL^T), fp32 WMMA 16x16x4. (Logits are O(1): no max-shift.)
__global__ void k_sim2(const float* QLf, const float* KLf, float* E) {
  extern __shared__ __align__(16) char smem[];
  float* QLs = (float*)smem;
  float* KLs = QLs + 256 * 64;
  int bh = blockIdx.x, tid = threadIdx.x, lane = tid & 31, w = tid >> 5;
  async_copy_contig<4096>(QLs, QLf + (size_t)bh * 256 * 64, tid);
  async_copy_contig<4096>(KLs, KLf + (size_t)bh * 256 * 64, tid);
  wait_async0();
  __syncthreads();
  int wr = w * 32;
  float* outg = E + (size_t)bh * 65536;
  for (int ch = 0; ch < 2; ++ch) {
    v8f acc[2][8] = {};
#pragma unroll
    for (int k = 0; k < 64; k += 4) {
      v2f a0 = load_a_f32(QLs + (wr + 0) * 64 + k, 64, lane);
      v2f a1 = load_a_f32(QLs + (wr + 16) * 64 + k, 64, lane);
#pragma unroll
      for (int j = 0; j < 8; ++j) {
        v2f b = load_bt_f32(KLs + (ch * 128 + j * 16) * 64 + k, 64, lane);
        acc[0][j] = wmma_f32(a0, b, acc[0][j]);
        acc[1][j] = wmma_f32(a1, b, acc[1][j]);
      }
    }
    int col = lane & 15, h = (lane >> 4) & 1;
#pragma unroll
    for (int i = 0; i < 2; ++i)
#pragma unroll
      for (int j = 0; j < 8; ++j) {
        int c0 = ch * 128 + j * 16 + col;
#pragma unroll
        for (int e = 0; e < 8; ++e)
          outg[(wr + i * 16 + 8 * h + e) * 256 + c0] = expf(acc[i][j][e]);
      }
  }
}

// K5b/c/d: normalize rows; column-sum max; z0 = attn2^T / rmax.
__global__ void k_rownorm(float* A2) {
  int row = blockIdx.x, t = threadIdx.x;
  float* a = A2 + (size_t)row * 256;
  __shared__ float red[256];
  float v = a[t];
  red[t] = v; __syncthreads();
  for (int o = 128; o > 0; o >>= 1) { if (t < o) red[t] += red[t + o]; __syncthreads(); }
  a[t] = v / red[0];
}
__global__ void k_colmax(const float* A2, float* RMAX) {
  int bh = blockIdx.x, t = threadIdx.x;
  const float* a = A2 + (size_t)bh * 65536;
  float cs = 0.f;
  for (int r = 0; r < 256; ++r) cs += a[r * 256 + t];
  __shared__ float red[256];
  red[t] = cs; __syncthreads();
  for (int o = 128; o > 0; o >>= 1) { if (t < o) red[t] = fmaxf(red[t], red[t + o]); __syncthreads(); }
  if (t == 0) RMAX[bh] = red[0];
}
__global__ void k_z0(const float* A2, const float* RMAX, float* Z) {
  int bh = blockIdx.x >> 8, j = blockIdx.x & 255, t = threadIdx.x;
  float r = RMAX[bh];
  Z[(size_t)bh * 65536 + j * 256 + t] = A2[(size_t)bh * 65536 + t * 256 + j] / r;
}

// K6: batched fp32 WMMA matmul  D = s*(A@B) + t*A.  A async; B transposed (34).
__global__ void k_pinv_mm(const float* A, const float* Bm, float* D, float s, float t) {
  __shared__ __align__(16) float As[128 * 32];
  __shared__ __align__(16) float Bst[128 * 34];
  int bh = blockIdx.z;
  int gm0 = blockIdx.y * 128, gn0 = blockIdx.x * 128;
  const float* Ag = A + (size_t)bh * 65536;
  const float* Bg = Bm + (size_t)bh * 65536;
  float* Dg = D + (size_t)bh * 65536;
  int tid = threadIdx.x, lane = tid & 31, w = tid >> 5;
  int wm = (w & 3) * 32, wn = (w >> 2) * 64;
  v8f acc[2][4] = {};
  for (int kk = 0; kk < 256; kk += 32) {
    async_copy_rows<128, 8>(As, (const char*)(Ag + (size_t)gm0 * 256 + kk), 1024, tid);
#pragma unroll
    for (int it = 0; it < 2; ++it) {  // 16 k-pairs x 32 n-groups(4)
      int u = tid + it * 256;
      int kp = u >> 5, ng = u & 31;
      const float* g0 = Bg + (size_t)(kk + 2 * kp) * 256 + gn0 + ng * 4;
      float4 ra = *(const float4*)g0, rb = *(const float4*)(g0 + 256);
      float av[4] = {ra.x, ra.y, ra.z, ra.w}, bv[4] = {rb.x, rb.y, rb.z, rb.w};
#pragma unroll
      for (int nn = 0; nn < 4; ++nn) {
        float2 pr; pr.x = av[nn]; pr.y = bv[nn];
        *(float2*)(Bst + (size_t)(ng * 4 + nn) * 34 + 2 * kp) = pr;
      }
    }
    wait_async0();
    __syncthreads();
#pragma unroll
    for (int k = 0; k < 32; k += 4) {
      v2f a0 = load_a_f32(As + (wm + 0) * 32 + k, 32, lane);
      v2f a1 = load_a_f32(As + (wm + 16) * 32 + k, 32, lane);
#pragma unroll
      for (int j = 0; j < 4; ++j) {
        v2f b = load_bt_f32(Bst + (wn + j * 16) * 34 + k, 34, lane);
        acc[0][j] = wmma_f32(a0, b, acc[0][j]);
        acc[1][j] = wmma_f32(a1, b, acc[1][j]);
      }
    }
    __syncthreads();
  }
  int col = lane & 15, h = (lane >> 4) & 1;
#pragma unroll
  for (int i = 0; i < 2; ++i)
#pragma unroll
    for (int j = 0; j < 4; ++j) {
      int c0 = gn0 + wn + j * 16 + col;
#pragma unroll
      for (int e = 0; e < 8; ++e) {
        int r0 = gm0 + wm + i * 16 + 8 * h + e;
        float v = s * acc[i][j][e];
        if (t != 0.f) v += t * Ag[(size_t)r0 * 256 + c0];
        Dg[(size_t)r0 * 256 + c0] = v;
      }
    }
}

// K7: S3V = softmax(QL @ K^T) @ V, streaming N in 128-row tiles; async K staging.
__global__ void k_attn3v(const u16* QLb, const u16* Kb, const u16* Vb, float* S3V) {
  extern __shared__ __align__(16) char smem[];
  u16* QLs = (u16*)smem;            // 256*64
  u16* Ks = QLs + 256 * 64;         // 128*64  (row-major, async-staged)
  u16* Vst = Ks + 128 * 64;         // 64*136  (transposed Vst[d][k])
  u16* Ps = Vst + 64 * 136;         // 256*128
  float* dls = (float*)(Ps + 256 * 128);
  int bh = blockIdx.x, tid = threadIdx.x, lane = tid & 31, w = tid >> 5;
  int wr = w * 32;
  async_copy_contig<2048>(QLs, QLb + (size_t)bh * 256 * 64, tid);
  wait_async0();
  __syncthreads();
  v16bf qa[2][2];
#pragma unroll
  for (int kk2 = 0; kk2 < 2; ++kk2) {
    qa[kk2][0] = load_a_bf16(QLs + (wr + 0) * 64 + kk2 * 32, 64, lane);
    qa[kk2][1] = load_a_bf16(QLs + (wr + 16) * 64 + kk2 * 32, 64, lane);
  }
  float denom = 0.f;
  v8f oacc[2][4] = {};
  for (int c = 0; c < 64; ++c) {
    int n0 = c * 128;
    // K tile: 128x64 bf16 row-major via async global->LDS DMA.
    async_copy_contig<1024>(Ks, Kb + ((size_t)bh * 8192 + n0) * 64, tid);
    // V tile transposed into Vst[d][k] (pairs of positions packed per b32).
    {
      const u16* vg = Vb + ((size_t)bh * 8192 + n0) * 64;
#pragma unroll
      for (int it = 0; it < 2; ++it) {  // 64 pos-pairs x 8 dim-groups
        int u = tid + it * 256;
        int kp = u >> 3, dg = u & 7;
        const u16* g0 = vg + (size_t)(2 * kp) * 64 + dg * 8;
        uint4 ra = *(const uint4*)g0, rb = *(const uint4*)(g0 + 64);
        uint32_t av[4] = {ra.x, ra.y, ra.z, ra.w}, bv[4] = {rb.x, rb.y, rb.z, rb.w};
#pragma unroll
        for (int nn = 0; nn < 8; ++nn)
          *(uint32_t*)(Vst + (dg * 8 + nn) * 136 + 2 * kp) = pack_pair(av[nn >> 1], bv[nn >> 1], nn & 1);
      }
    }
    wait_async0();
    __syncthreads();
    // S chunk = QL @ K^T ; exp -> Ps (bf16)
#pragma unroll
    for (int jj = 0; jj < 8; ++jj) {
      v8f s0 = {}, s1 = {};
#pragma unroll
      for (int kk2 = 0; kk2 < 2; ++kk2) {
        v16bf b = load_bt_bf16(Ks + (jj * 16) * 64 + kk2 * 32, 64, lane);
        s0 = wmma_bf16(qa[kk2][0], b, s0);
        s1 = wmma_bf16(qa[kk2][1], b, s1);
      }
      int col = jj * 16 + (lane & 15), h = (lane >> 4) & 1;
#pragma unroll
      for (int e = 0; e < 8; ++e) {
        Ps[(wr + 8 * h + e) * 128 + col] = f2bf(expf(s0[e]));
        Ps[(wr + 16 + 8 * h + e) * 128 + col] = f2bf(expf(s1[e]));
      }
    }
    __syncthreads();
    {  // denominator: thread t owns row t
      const uint4* pr = (const uint4*)(Ps + tid * 128);
#pragma unroll
      for (int xx = 0; xx < 16; ++xx) denom += sum8bf(pr[xx]);
    }
    // O += P @ V
#pragma unroll
    for (int kk = 0; kk < 128; kk += 32) {
      v16bf a0 = load_a_bf16(Ps + (wr + 0) * 128 + kk, 128, lane);
      v16bf a1 = load_a_bf16(Ps + (wr + 16) * 128 + kk, 128, lane);
#pragma unroll
      for (int j = 0; j < 4; ++j) {
        v16bf b = load_bt_bf16(Vst + (j * 16) * 136 + kk, 136, lane);
        oacc[0][j] = wmma_bf16(a0, b, oacc[0][j]);
        oacc[1][j] = wmma_bf16(a1, b, oacc[1][j]);
      }
    }
    __syncthreads();
  }
  dls[tid] = denom; __syncthreads();
  int col = lane & 15, h = (lane >> 4) & 1;
  float* og = S3V + (size_t)bh * 256 * 64;
#pragma unroll
  for (int i = 0; i < 2; ++i)
#pragma unroll
    for (int j = 0; j < 4; ++j)
#pragma unroll
      for (int e = 0; e < 8; ++e) {
        int r = wr + i * 16 + 8 * h + e;
        og[r * 64 + j * 16 + col] = oacc[i][j][e] / dls[r];
      }
}

// K8: M = Z_final @ S3V, fp32 WMMA -> bf16. A async; B transposed (34).
__global__ void k_mm_zs(const float* Z, const float* S3V, u16* Mb) {
  __shared__ __align__(16) float As[256 * 32];
  __shared__ __align__(16) float Bst[64 * 34];
  int bh = blockIdx.x, tid = threadIdx.x, lane = tid & 31, w = tid >> 5;
  int wr = w * 32;
  const float* zg = Z + (size_t)bh * 65536;
  const float* bg = S3V + (size_t)bh * 16384;
  v8f acc[2][4] = {};
  for (int kk = 0; kk < 256; kk += 32) {
    async_copy_rows<256, 8>(As, (const char*)(zg + kk), 1024, tid);
    {  // B: 32x64, transpose-staged
      int kp = tid >> 4, ng = tid & 15;
      const float* g0 = bg + (size_t)(kk + 2 * kp) * 64 + ng * 4;
      float4 ra = *(const float4*)g0, rb = *(const float4*)(g0 + 64);
      float av[4] = {ra.x, ra.y, ra.z, ra.w}, bv[4] = {rb.x, rb.y, rb.z, rb.w};
#pragma unroll
      for (int nn = 0; nn < 4; ++nn) {
        float2 pr; pr.x = av[nn]; pr.y = bv[nn];
        *(float2*)(Bst + (size_t)(ng * 4 + nn) * 34 + 2 * kp) = pr;
      }
    }
    wait_async0();
    __syncthreads();
#pragma unroll
    for (int k = 0; k < 32; k += 4) {
      v2f a0 = load_a_f32(As + (wr + 0) * 32 + k, 32, lane);
      v2f a1 = load_a_f32(As + (wr + 16) * 32 + k, 32, lane);
#pragma unroll
      for (int j = 0; j < 4; ++j) {
        v2f b = load_bt_f32(Bst + (j * 16) * 34 + k, 34, lane);
        acc[0][j] = wmma_f32(a0, b, acc[0][j]);
        acc[1][j] = wmma_f32(a1, b, acc[1][j]);
      }
    }
    __syncthreads();
  }
  u16* mg = Mb + (size_t)bh * 16384;
  int col = lane & 15, h = (lane >> 4) & 1;
#pragma unroll
  for (int i = 0; i < 2; ++i)
#pragma unroll
    for (int j = 0; j < 4; ++j)
#pragma unroll
      for (int e = 0; e < 8; ++e)
        mg[(wr + i * 16 + 8 * h + e) * 64 + j * 16 + col] = f2bf(acc[i][j][e]);
}

// K9: depthwise 33-tap conv over sequence on v (async staging + vector rows).
__global__ void k_conv(const u16* Vb, const float* rk, float* RES) {
  __shared__ __align__(16) u16 vs[160 * 64];
  __shared__ float wk[33];
  int bh = blockIdx.y, tile = blockIdx.x, n0 = tile * 128;
  int hh = bh & 7, tid = threadIdx.x;
  if (tid < 33) wk[tid] = rk[hh * 33 + tid];
  const u16* vg = Vb + (size_t)bh * 8192 * 64;
  uint32_t vs_base = (uint32_t)(uintptr_t)vs;
#pragma unroll
  for (int it = 0; it < 5; ++it) {  // 1280 x 16B chunks
    int i = tid + it * 256;
    int r = i >> 3, c8 = i & 7;
    int n = n0 - 16 + r;
    if (n >= 0 && n < 8192) {
      async_ld_b128(vs_base + i * 16, vg + (size_t)n * 64 + c8 * 8);
    } else {
      uint4 z = make_uint4(0, 0, 0, 0);
      *(uint4*)(vs + i * 8) = z;
    }
  }
  wait_async0();
  __syncthreads();
#pragma unroll
  for (int it = 0; it < 4; ++it) {  // 128 positions x 8 dim-groups
    int u = tid + it * 256;
    int r = u >> 3, dg = u & 7;
    float a[8] = {};
#pragma unroll
    for (int t = 0; t < 33; ++t) {
      uint4 v = *(const uint4*)(vs + (r + t) * 64 + dg * 8);
      float wt = wk[t];
      uint32_t wv[4] = {v.x, v.y, v.z, v.w};
#pragma unroll
      for (int q = 0; q < 4; ++q) {
        a[2 * q] += wt * bf2f((u16)(wv[q] & 0xffff));
        a[2 * q + 1] += wt * bf2f((u16)(wv[q] >> 16));
      }
    }
    float* og = RES + ((size_t)bh * 8192 + n0 + r) * 64 + dg * 8;
    *(float4*)og = make_float4(a[0], a[1], a[2], a[3]);
    *(float4*)(og + 4) = make_float4(a[4], a[5], a[6], a[7]);
  }
}

// K10: OUT = softmax(Q @ KL^T) @ M + RES -> merged-head bf16.
__global__ void k_attn1c(const u16* Qb, const u16* KLb, const u16* Mb, const float* RES, u16* OUTH) {
  extern __shared__ __align__(16) char smem[];
  u16* Qs = (u16*)smem;             // 256*64
  u16* KLs = Qs + 256 * 64;         // 256*64
  u16* Mst = KLs + 256 * 64;        // 64*264 (transposed Mst[d][j])
  u16* Ps = Mst + 64 * 264;         // 256*64
  float* dls = (float*)(Ps + 256 * 64);
  int bh = blockIdx.y, t0 = blockIdx.x * 256;
  int tid = threadIdx.x, lane = tid & 31, w = tid >> 5, wr = w * 32;
  async_copy_contig<2048>(Qs, Qb + ((size_t)bh * 8192 + t0) * 64, tid);
  async_copy_contig<2048>(KLs, KLb + (size_t)bh * 256 * 64, tid);
  {  // M (256x64) -> Mst[d][j]
    const u16* mg = Mb + (size_t)bh * 256 * 64;
#pragma unroll
    for (int it = 0; it < 4; ++it) {  // 128 j-pairs x 8 dim-groups
      int u = tid + it * 256;
      int jp = u >> 3, dg = u & 7;
      const u16* g0 = mg + (size_t)(2 * jp) * 64 + dg * 8;
      uint4 ra = *(const uint4*)g0, rb = *(const uint4*)(g0 + 64);
      uint32_t av[4] = {ra.x, ra.y, ra.z, ra.w}, bv[4] = {rb.x, rb.y, rb.z, rb.w};
#pragma unroll
      for (int nn = 0; nn < 8; ++nn)
        *(uint32_t*)(Mst + (dg * 8 + nn) * 264 + 2 * jp) = pack_pair(av[nn >> 1], bv[nn >> 1], nn & 1);
    }
  }
  wait_async0();
  __syncthreads();
  v16bf qa[2][2];
#pragma unroll
  for (int kk2 = 0; kk2 < 2; ++kk2) {
    qa[kk2][0] = load_a_bf16(Qs + (wr + 0) * 64 + kk2 * 32, 64, lane);
    qa[kk2][1] = load_a_bf16(Qs + (wr + 16) * 64 + kk2 * 32, 64, lane);
  }
  float denom = 0.f;
  v8f oacc[2][4] = {};
  for (int jc = 0; jc < 256; jc += 64) {
#pragma unroll
    for (int jj = 0; jj < 4; ++jj) {
      v8f s0 = {}, s1 = {};
#pragma unroll
      for (int kk2 = 0; kk2 < 2; ++kk2) {
        v16bf b = load_bt_bf16(KLs + (jc + jj * 16) * 64 + kk2 * 32, 64, lane);
        s0 = wmma_bf16(qa[kk2][0], b, s0);
        s1 = wmma_bf16(qa[kk2][1], b, s1);
      }
      int col = jj * 16 + (lane & 15), h = (lane >> 4) & 1;
#pragma unroll
      for (int e = 0; e < 8; ++e) {
        Ps[(wr + 8 * h + e) * 64 + col] = f2bf(expf(s0[e]));
        Ps[(wr + 16 + 8 * h + e) * 64 + col] = f2bf(expf(s1[e]));
      }
    }
    __syncthreads();
    {
      const uint4* pr = (const uint4*)(Ps + tid * 64);
#pragma unroll
      for (int xx = 0; xx < 8; ++xx) denom += sum8bf(pr[xx]);
    }
#pragma unroll
    for (int kk = 0; kk < 64; kk += 32) {
      v16bf a0 = load_a_bf16(Ps + (wr + 0) * 64 + kk, 64, lane);
      v16bf a1 = load_a_bf16(Ps + (wr + 16) * 64 + kk, 64, lane);
#pragma unroll
      for (int j = 0; j < 4; ++j) {
        v16bf b = load_bt_bf16(Mst + (j * 16) * 264 + jc + kk, 264, lane);
        oacc[0][j] = wmma_bf16(a0, b, oacc[0][j]);
        oacc[1][j] = wmma_bf16(a1, b, oacc[1][j]);
      }
    }
    __syncthreads();
  }
  dls[tid] = denom; __syncthreads();
  int col = lane & 15, h = (lane >> 4) & 1;
  int bidx = bh >> 3, hh = bh & 7;
#pragma unroll
  for (int i = 0; i < 2; ++i)
#pragma unroll
    for (int j = 0; j < 4; ++j)
#pragma unroll
      for (int e = 0; e < 8; ++e) {
        int r = wr + i * 16 + 8 * h + e;
        int n = t0 + r;
        float v = oacc[i][j][e] / dls[r] + RES[((size_t)bh * 8192 + n) * 64 + j * 16 + col];
        OUTH[((size_t)bidx * 8192 + n) * 512 + hh * 64 + j * 16 + col] = f2bf(v);
      }
}

// K11: Y = OUTH @ Wout + b_out + x, bf16 WMMA; A async; B transposed.
__global__ void k_gemm_out(const u16* OUTH, const u16* Wb, const float* bias, const float* x, float* out) {
  __shared__ __align__(16) u16 As[128 * 32];
  __shared__ __align__(16) u16 Bst[128 * 40];
  int gm0 = blockIdx.y * 128, gn0 = blockIdx.x * 128;
  int tid = threadIdx.x, lane = tid & 31, w = tid >> 5;
  int wm = (w & 3) * 32, wn = (w >> 2) * 64;
  v8f acc[2][4] = {};
  for (int kk = 0; kk < 512; kk += 32) {
    async_copy_rows<128, 4>(As, (const char*)(OUTH + (size_t)gm0 * 512 + kk), 1024, tid);
    {
      int kp = tid >> 4, ng = tid & 15;
      const u16* g0 = Wb + (size_t)(kk + 2 * kp) * 512 + gn0 + ng * 8;
      uint4 ra = *(const uint4*)g0, rb = *(const uint4*)(g0 + 512);
      uint32_t av[4] = {ra.x, ra.y, ra.z, ra.w}, bv[4] = {rb.x, rb.y, rb.z, rb.w};
#pragma unroll
      for (int nn = 0; nn < 8; ++nn)
        *(uint32_t*)(Bst + (ng * 8 + nn) * 40 + 2 * kp) = pack_pair(av[nn >> 1], bv[nn >> 1], nn & 1);
    }
    wait_async0();
    __syncthreads();
    v16bf a0 = load_a_bf16(As + (wm + 0) * 32, 32, lane);
    v16bf a1 = load_a_bf16(As + (wm + 16) * 32, 32, lane);
#pragma unroll
    for (int j = 0; j < 4; ++j) {
      v16bf b = load_bt_bf16(Bst + (wn + j * 16) * 40, 40, lane);
      acc[0][j] = wmma_bf16(a0, b, acc[0][j]);
      acc[1][j] = wmma_bf16(a1, b, acc[1][j]);
    }
    __syncthreads();
  }
  int col = lane & 15, h = (lane >> 4) & 1;
#pragma unroll
  for (int i = 0; i < 2; ++i)
#pragma unroll
    for (int j = 0; j < 4; ++j) {
      int gc = gn0 + wn + j * 16 + col;
#pragma unroll
      for (int e = 0; e < 8; ++e) {
        int gr = gm0 + wm + i * 16 + 8 * h + e;
        out[(size_t)gr * 512 + gc] = acc[i][j][e] + bias[gc] + x[(size_t)gr * 512 + gc];
      }
    }
}

// =============================================================================
extern "C" void kernel_launch(void* const* d_in, const int* in_sizes, int n_in,
                              void* d_out, int out_size, void* d_ws, size_t ws_size,
                              hipStream_t stream) {
  (void)in_sizes; (void)n_in; (void)out_size; (void)ws_size;
  const float* x = (const float*)d_in[0];
  const float* lng = (const float*)d_in[1];
  const float* lnb = (const float*)d_in[2];
  const float* wqkv = (const float*)d_in[3];
  const float* wout = (const float*)d_in[4];
  const float* bout = (const float*)d_in[5];
  const float* rk = (const float*)d_in[6];
  float* out = (float*)d_out;

  char* base = (char*)d_ws;
  size_t off = 0;
  auto take = [&](size_t bytes) -> char* {
    char* r = base + off;
    off = (off + bytes + 255) & ~(size_t)255;
    return r;
  };
  u16* XN = (u16*)take(32768ull * 512 * 2);
  u16* WQKVb = (u16*)take(512ull * 1536 * 2);
  u16* WOUTb = (u16*)take(512ull * 512 * 2);
  u16* Qb = (u16*)take(32ull * 8192 * 64 * 2);
  u16* Kb = (u16*)take(32ull * 8192 * 64 * 2);
  u16* Vb = (u16*)take(32ull * 8192 * 64 * 2);
  float* QLf = (float*)take(32ull * 256 * 64 * 4);
  float* KLf = (float*)take(32ull * 256 * 64 * 4);
  u16* QLb = (u16*)take(32ull * 256 * 64 * 2);
  u16* KLb = (u16*)take(32ull * 256 * 64 * 2);
  float* A2 = (float*)take(32ull * 65536 * 4);
  float* ZA = (float*)take(32ull * 65536 * 4);
  float* ZB = (float*)take(32ull * 65536 * 4);
  float* XZ = (float*)take(32ull * 65536 * 4);
  float* W1 = (float*)take(32ull * 65536 * 4);
  float* W2 = (float*)take(32ull * 65536 * 4);
  float* RMAX = (float*)take(32 * 4);
  float* S3V = (float*)take(32ull * 16384 * 4);
  u16* Mb = (u16*)take(32ull * 16384 * 2);
  float* RES = (float*)take(32ull * 8192 * 64 * 4);
  u16* OUTH = (u16*)take(32768ull * 512 * 2);

  (void)hipFuncSetAttribute((const void*)k_sim2, hipFuncAttributeMaxDynamicSharedMemorySize, 131072);
  (void)hipFuncSetAttribute((const void*)k_attn3v, hipFuncAttributeMaxDynamicSharedMemorySize, 133120);
  (void)hipFuncSetAttribute((const void*)k_attn1c, hipFuncAttributeMaxDynamicSharedMemorySize, 133120);

  k_cvt_weights<<<1536, 256, 0, stream>>>(wqkv, wout, WQKVb, WOUTb);
  k_layernorm<<<32768, 256, 0, stream>>>(x, lng, lnb, XN);
  k_gemm_qkv<<<dim3(12, 256), 256, 0, stream>>>(XN, WQKVb, Qb, Kb, Vb);
  k_landmarks<<<8192, 64, 0, stream>>>(Qb, Kb, QLf, KLf, QLb, KLb);
  k_sim2<<<32, 256, 131072, stream>>>(QLf, KLf, A2);
  k_rownorm<<<8192, 256, 0, stream>>>(A2);
  k_colmax<<<32, 256, 0, stream>>>(A2, RMAX);
  k_z0<<<8192, 256, 0, stream>>>(A2, RMAX, ZA);

  // Moore-Penrose Newton iteration: z <- 0.25*z*(13I - xz*(15I - xz*(7I - xz)))
  float* zc = ZA; float* zn = ZB;
  for (int it = 0; it < 6; ++it) {
    k_pinv_mm<<<dim3(2, 2, 32), 256, 0, stream>>>(A2, zc, XZ, 1.f, 0.f);
    k_pinv_mm<<<dim3(2, 2, 32), 256, 0, stream>>>(XZ, XZ, W1, -1.f, 7.f);
    k_pinv_mm<<<dim3(2, 2, 32), 256, 0, stream>>>(XZ, W1, W2, -1.f, 15.f);
    k_pinv_mm<<<dim3(2, 2, 32), 256, 0, stream>>>(zc, W2, zn, -0.25f, 3.25f);
    float* t = zc; zc = zn; zn = t;
  }

  k_attn3v<<<32, 256, 133120, stream>>>(QLb, Kb, Vb, S3V);
  k_mm_zs<<<32, 256, 0, stream>>>(zc, S3V, Mb);
  k_conv<<<dim3(64, 32), 256, 0, stream>>>(Vb, rk, RES);
  k_attn1c<<<dim3(32, 32), 256, 133120, stream>>>(Qb, KLb, Mb, RES, OUTH);
  k_gemm_out<<<dim3(4, 256), 256, 0, stream>>>(OUTH, WOUTb, bout, x, out);
}
